// SpikingBlock_51719996178452
// MI455X (gfx1250) — compile-verified, matrix operand order checked by reading, not run on hardware
//
#include <hip/hip_runtime.h>
#include <hip/hip_bf16.h>
#include <stdint.h>

#define Bsz   2
#define Ssz   2048
#define Dsz   768
#define FFsz  3072
#define NHsz  12
#define HD    64
#define NROWS (Bsz * Ssz)    // 4096
#define QKVLD (3 * Dsz)      // 2304, fused qkv row stride

typedef __attribute__((ext_vector_type(16))) __bf16 v16bf;
typedef __attribute__((ext_vector_type(8)))  __bf16 v8bf;
typedef __attribute__((ext_vector_type(8)))  float  v8f;

// ---------------------------------------------------------------- WMMA helpers

__device__ __forceinline__ v8f wmma_bf16(v16bf a, v16bf b, v8f c) {
  // D = A(16x32 bf16) * B(32x16 bf16) + C(16x16 f32)
  return __builtin_amdgcn_wmma_f32_16x16x32_bf16(
      /*neg_a=*/false, a, /*neg_b=*/false, b,
      /*c_mod=*/(short)0, c, /*reuse_a=*/false, /*reuse_b=*/false);
}

// Load a 16x32 bf16 A/B fragment from row-major storage (ld in elements).
// CDNA5 layout: lane L holds row (L&15); 16 halves = K[off..off+8) U K[off+16..off+24),
// off = (L>>4)*8.  Two 16B vector loads per lane.
__device__ __forceinline__ v16bf load_frag(const __bf16* base, int ld) {
  const int lane = threadIdx.x & 31;
  const __bf16* p = base + (lane & 15) * ld + ((lane >> 4) << 3);
  union { v16bf v; v8bf h[2]; } u;
  u.h[0] = *(const v8bf*)(p);
  u.h[1] = *(const v8bf*)(p + 16);
  return u.v;
}

// ---------------------------------------------------------------- small kernels

__global__ __launch_bounds__(256)
void f2bf_kernel(const float* __restrict__ in, __bf16* __restrict__ out, int n) {
  int i = blockIdx.x * 256 + threadIdx.x;
  if (i < n) out[i] = (__bf16)in[i];
}

// y = rmsnorm(x) * w, cast to bf16.  One block (256 thr) per row of D=768.
__global__ __launch_bounds__(256)
void rmsnorm_cast(const float* __restrict__ X, const float* __restrict__ W,
                  __bf16* __restrict__ Y) {
  const int row = blockIdx.x;
  const float* xr = X + (size_t)row * Dsz;
  float ss = 0.f;
  for (int i = threadIdx.x; i < Dsz; i += 256) { float v = xr[i]; ss += v * v; }
#pragma unroll
  for (int msk = 16; msk >= 1; msk >>= 1) ss += __shfl_xor(ss, msk);
  __shared__ float red[8];
  if ((threadIdx.x & 31) == 0) red[threadIdx.x >> 5] = ss;
  __syncthreads();
  if (threadIdx.x == 0) {
    float t = 0.f;
#pragma unroll
    for (int w = 0; w < 8; ++w) t += red[w];
    red[0] = t;
  }
  __syncthreads();
  const float rinv = rsqrtf(red[0] * (1.0f / Dsz) + 1e-6f);
  for (int i = threadIdx.x; i < Dsz; i += 256)
    Y[(size_t)row * Dsz + i] = (__bf16)(xr[i] * rinv * W[i]);
}

// ---------------------------------------------------------------- tiled GEMM
// C[M,N] = A[M,K] * B[N,K]^T  (both K-major).  Double-buffered LDS, one
// barrier per 32-wide K-step; next tile is prefetched into registers while
// the current tile feeds 8 WMMAs per wave.
// Block: 256 thr = 8 waves in a 4(M) x 2(N) grid; tile 128x128.
// MODE 0: Cf = acc + addend (f32)      MODE 1: Cb = bf16(acc)
// MODE 2: Cb = bf16( LIF(acc, beta[col]) )   (T=4 leaky integrate-and-fire)
template <int MODE>
__global__ __launch_bounds__(256)
void gemm_bf16(const __bf16* __restrict__ A, const __bf16* __restrict__ Bm,
               int M, int N, int K,
               float* __restrict__ Cf, __bf16* __restrict__ Cb,
               const float* __restrict__ addend, const float* __restrict__ beta) {
  __shared__ __bf16 As[2][128 * 32];
  __shared__ __bf16 Bs[2][128 * 32];
  const int tid  = threadIdx.x;
  const int wid  = tid >> 5;
  const int lane = tid & 31;
  const int bm = blockIdx.y * 128, bn = blockIdx.x * 128;
  const int wm = (wid >> 1) * 32;  // 0,32,64,96
  const int wn = (wid & 1) * 64;   // 0,64
  v8f acc[2][4] = {};
  const int lr = tid >> 1;        // 0..127 (tile row for cooperative loads)
  const int lc = (tid & 1) * 16;  // 0 / 16
  const size_t arow = (size_t)(bm + lr) * K + lc;
  const size_t brow = (size_t)(bn + lr) * K + lc;
  const int nk = K / 32;

  // prefetch tile 0 into regs, stage into LDS[0]
  v8bf a0 = *(const v8bf*)&A[arow];
  v8bf a1 = *(const v8bf*)&A[arow + 8];
  v8bf b0 = *(const v8bf*)&Bm[brow];
  v8bf b1 = *(const v8bf*)&Bm[brow + 8];
  *(v8bf*)&As[0][lr * 32 + lc]     = a0;
  *(v8bf*)&As[0][lr * 32 + lc + 8] = a1;
  *(v8bf*)&Bs[0][lr * 32 + lc]     = b0;
  *(v8bf*)&Bs[0][lr * 32 + lc + 8] = b1;
  __syncthreads();

  int buf = 0;
  for (int kt = 0; kt < nk; ++kt) {
    if (kt + 1 < nk) {  // prefetch next K-slab (overlaps with WMMA below)
      const int k0 = (kt + 1) * 32;
      a0 = *(const v8bf*)&A[arow + k0];
      a1 = *(const v8bf*)&A[arow + k0 + 8];
      b0 = *(const v8bf*)&Bm[brow + k0];
      b1 = *(const v8bf*)&Bm[brow + k0 + 8];
    }
    v16bf af[2], bfr[4];
#pragma unroll
    for (int mi = 0; mi < 2; ++mi) af[mi]  = load_frag(&As[buf][(wm + mi * 16) * 32], 32);
#pragma unroll
    for (int ni = 0; ni < 4; ++ni) bfr[ni] = load_frag(&Bs[buf][(wn + ni * 16) * 32], 32);
#pragma unroll
    for (int mi = 0; mi < 2; ++mi)
#pragma unroll
      for (int ni = 0; ni < 4; ++ni)
        acc[mi][ni] = wmma_bf16(af[mi], bfr[ni], acc[mi][ni]);
    if (kt + 1 < nk) {  // stage prefetched regs into the other buffer
      *(v8bf*)&As[buf ^ 1][lr * 32 + lc]     = a0;
      *(v8bf*)&As[buf ^ 1][lr * 32 + lc + 8] = a1;
      *(v8bf*)&Bs[buf ^ 1][lr * 32 + lc]     = b0;
      *(v8bf*)&Bs[buf ^ 1][lr * 32 + lc + 8] = b1;
    }
    __syncthreads();
    buf ^= 1;
  }

  const int cn   = lane & 15;
  const int radd = (lane >> 4) << 3;  // rows r (lanes 0-15) / r+8 (lanes 16-31)
#pragma unroll
  for (int mi = 0; mi < 2; ++mi)
#pragma unroll
    for (int ni = 0; ni < 4; ++ni) {
#pragma unroll
      for (int r = 0; r < 8; ++r) {
        const int row = bm + wm + mi * 16 + radd + r;
        const int col = bn + wn + ni * 16 + cn;
        const float v = acc[mi][ni][r];
        if (MODE == 0) {
          Cf[(size_t)row * N + col] = v + (addend ? addend[(size_t)row * N + col] : 0.f);
        } else if (MODE == 1) {
          Cb[(size_t)row * N + col] = (__bf16)v;
        } else {  // fused LIF rate, THRESHOLD=1, soft reset, T=4
          const float bq = beta[col];
          float vm = 0.f, ssum = 0.f;
#pragma unroll
          for (int t = 0; t < 4; ++t) {
            vm = bq * vm + v;
            const float s = (vm >= 1.0f) ? 1.0f : 0.0f;
            ssum += s;
            vm -= s;
          }
          Cb[(size_t)row * N + col] = (__bf16)(ssum * 0.25f);
        }
      }
    }
}

// ---------------------------------------------------------------- flash attention
// grid (S/64, B*NH), block 128 = 4 waves; each wave owns 16 q-rows.
// Q/K/V come from the fused qkv buffer (row stride QKVLD); output stride Dsz.
// Next kv-tile is prefetched into registers during the current tile's compute.
__global__ __launch_bounds__(128)
void flash_attn(const __bf16* __restrict__ Q, const __bf16* __restrict__ Kg,
                const __bf16* __restrict__ Vg, __bf16* __restrict__ Og) {
  __shared__ __bf16 Ks[64 * 64];       // K tile, row-major [kv][d]
  __shared__ __bf16 VTs[64 * 64];      // V tile transposed [d][kv]
  __shared__ __bf16 Ps[4][16 * 64];    // per-wave P relayout buffer
  const int tid = threadIdx.x, wid = tid >> 5, lane = tid & 31;
  const int qt = blockIdx.x, bh = blockIdx.y;
  const int b = bh / NHsz, h = bh % NHsz;
  const size_t inoff  = (size_t)b * Ssz * QKVLD + (size_t)h * HD;
  const size_t outoff = (size_t)b * Ssz * Dsz   + (size_t)h * HD;
  const int qbase = qt * 64;
  const int myq = qbase + wid * 16;
  const int cn = lane & 15;
  const int radd = (lane >> 4) << 3;

  v16bf qf[2];
  {
    const __bf16* qp = Q + inoff + (size_t)myq * QKVLD;
    qf[0] = load_frag(qp, QKVLD);
    qf[1] = load_frag(qp + 32, QKVLD);
  }
  float m_i[8], l_i[8];
#pragma unroll
  for (int r = 0; r < 8; ++r) { m_i[r] = -1e30f; l_i[r] = 0.f; }
  v8f of[4] = {};

  // per-thread prefetch registers: 4 iterations x (K vec + V vec)
  v8bf kreg[4], vreg[4];
  const int frr = tid >> 3;          // fetch row 0..15 (stepped by 16 per iter... see below)
  const int fcc = (tid & 7) * 8;     // fetch col
  // i = tid + it*128 -> rr = i>>3 = frr + it*16, cc = fcc
  {
    const int kv0 = 0;
#pragma unroll
    for (int it = 0; it < 4; ++it) {
      const int rr = frr + it * 16;
      kreg[it] = *(const v8bf*)&Kg[inoff + (size_t)(kv0 + rr) * QKVLD + fcc];
      vreg[it] = *(const v8bf*)&Vg[inoff + (size_t)(kv0 + rr) * QKVLD + fcc];
    }
  }

  for (int kt = 0; kt <= qt; ++kt) {
    const int kv0 = kt * 64;
    // stage prefetched tile: K straight, V transposed
#pragma unroll
    for (int it = 0; it < 4; ++it) {
      const int rr = frr + it * 16;
      *(v8bf*)&Ks[rr * 64 + fcc] = kreg[it];
#pragma unroll
      for (int e = 0; e < 8; ++e) VTs[(fcc + e) * 64 + rr] = vreg[it][e];
    }
    __syncthreads();
    if (kt < qt) {  // prefetch next tile (overlaps with compute below)
      const int nv0 = kv0 + 64;
#pragma unroll
      for (int it = 0; it < 4; ++it) {
        const int rr = frr + it * 16;
        kreg[it] = *(const v8bf*)&Kg[inoff + (size_t)(nv0 + rr) * QKVLD + fcc];
        vreg[it] = *(const v8bf*)&Vg[inoff + (size_t)(nv0 + rr) * QKVLD + fcc];
      }
    }

    // S = Q K^T (4 column fragments of 16)
    v8f sf[4];
#pragma unroll
    for (int ni = 0; ni < 4; ++ni) {
      v8f s = {};
      s = wmma_bf16(qf[0], load_frag(&Ks[(ni * 16) * 64 + 0],  64), s);
      s = wmma_bf16(qf[1], load_frag(&Ks[(ni * 16) * 64 + 32], 64), s);
      sf[ni] = s;
    }
    // scale, causal mask, per-row max (reduce across the 16-lane group)
    float mt[8];
#pragma unroll
    for (int r = 0; r < 8; ++r) {
      const int qrow = myq + radd + r;
      float mv = -1e30f;
#pragma unroll
      for (int ni = 0; ni < 4; ++ni) {
        float sv = sf[ni][r] * 0.125f;  // 1/sqrt(64)
        const int kvcol = kv0 + ni * 16 + cn;
        if (kvcol > qrow) sv = -1e30f;
        sf[ni][r] = sv;
        mv = fmaxf(mv, sv);
      }
#pragma unroll
      for (int msk = 8; msk >= 1; msk >>= 1) mv = fmaxf(mv, __shfl_xor(mv, msk));
      mt[r] = mv;
    }
    // online softmax update
#pragma unroll
    for (int r = 0; r < 8; ++r) {
      const float mnew = fmaxf(m_i[r], mt[r]);
      const float scale = __expf(m_i[r] - mnew);
      float rs = 0.f;
#pragma unroll
      for (int ni = 0; ni < 4; ++ni) {
        const float p = __expf(sf[ni][r] - mnew);
        sf[ni][r] = p;
        rs += p;
      }
#pragma unroll
      for (int msk = 8; msk >= 1; msk >>= 1) rs += __shfl_xor(rs, msk);
      l_i[r] = l_i[r] * scale + rs;
      m_i[r] = mnew;
#pragma unroll
      for (int f = 0; f < 4; ++f) of[f][r] *= scale;
    }
    // P: C-fragment layout -> LDS -> A-fragment layout (same-wave LDS is in-order)
    __bf16* pb = Ps[wid];
#pragma unroll
    for (int ni = 0; ni < 4; ++ni)
#pragma unroll
      for (int r = 0; r < 8; ++r)
        pb[(radd + r) * 64 + ni * 16 + cn] = (__bf16)sf[ni][r];
    const v16bf pf0 = load_frag(pb, 64);
    const v16bf pf1 = load_frag(pb + 32, 64);
    // O += P V  (B operand = VT rows are V columns, contiguous)
#pragma unroll
    for (int f = 0; f < 4; ++f) {
      of[f] = wmma_bf16(pf0, load_frag(&VTs[(f * 16) * 64 + 0],  64), of[f]);
      of[f] = wmma_bf16(pf1, load_frag(&VTs[(f * 16) * 64 + 32], 64), of[f]);
    }
    __syncthreads();  // before next tile overwrites Ks/VTs
  }
  // normalize and store bf16 into o_buf [N, D]
#pragma unroll
  for (int r = 0; r < 8; ++r) {
    const float inv = 1.0f / l_i[r];
    const int row = myq + radd + r;
#pragma unroll
    for (int f = 0; f < 4; ++f)
      Og[outoff + (size_t)row * Dsz + f * 16 + cn] = (__bf16)(of[f][r] * inv);
  }
}

// ---------------------------------------------------------------- rate_per_unit
__global__ __launch_bounds__(256)
void col_partial(const __bf16* __restrict__ rate, float* __restrict__ partial,
                 int rows_per, int FF) {
  const int j = blockIdx.x * 256 + threadIdx.x;
  const int r0 = blockIdx.y * rows_per;
  float s = 0.f;
  for (int r = 0; r < rows_per; ++r) s += (float)rate[(size_t)(r0 + r) * FF + j];
  partial[(size_t)blockIdx.y * FF + j] = s;
}

__global__ __launch_bounds__(256)
void col_final(const float* __restrict__ partial, float* __restrict__ outv,
               int chunks, int FF, float invN) {
  const int j = blockIdx.x * 256 + threadIdx.x;
  float s = 0.f;
  for (int c = 0; c < chunks; ++c) s += partial[(size_t)c * FF + j];
  outv[j] = s * invN;
}

// ---------------------------------------------------------------- launch

extern "C" void kernel_launch(void* const* d_in, const int* in_sizes, int n_in,
                              void* d_out, int out_size, void* d_ws, size_t ws_size,
                              hipStream_t stream) {
  (void)in_sizes; (void)n_in; (void)out_size; (void)ws_size;
  const float* h    = (const float*)d_in[0];
  const float* anw  = (const float*)d_in[1];
  const float* wq   = (const float*)d_in[2];
  const float* wk   = (const float*)d_in[3];
  const float* wv   = (const float*)d_in[4];
  const float* wo   = (const float*)d_in[5];
  const float* mnw  = (const float*)d_in[6];
  const float* wup  = (const float*)d_in[7];
  const float* wdn  = (const float*)d_in[8];
  const float* beta = (const float*)d_in[9];
  float* out = (float*)d_out;  // [N*D] out, then [FF] rate_per_unit

  const size_t DD  = (size_t)Dsz * Dsz;
  const size_t DFF = (size_t)Dsz * FFsz;
  const size_t ND  = (size_t)NROWS * Dsz;
  const size_t NFF = (size_t)NROWS * FFsz;

  char* ws = (char*)d_ws;
  size_t off = 0;
  auto carve = [&](size_t bytes) -> void* {
    void* p = ws + off;
    off = (off + bytes + 255) & ~(size_t)255;
    return p;
  };
  __bf16* wqkv_b = (__bf16*)carve(3 * DD * 2);  // [2304, 768] concat q|k|v weights
  __bf16* wo_b   = (__bf16*)carve(DD * 2);
  __bf16* wup_b  = (__bf16*)carve(DFF * 2);
  __bf16* wdn_b  = (__bf16*)carve(DFF * 2);
  __bf16* x_b    = (__bf16*)carve(ND * 2);      // reused for x1 then x2
  __bf16* qkv_b  = (__bf16*)carve(3 * ND * 2);  // [4096, 2304] fused q|k|v
  __bf16* o_b    = (__bf16*)carve(ND * 2);
  float*  h1     = (float*)carve(ND * 4);
  __bf16* rate_b = (__bf16*)carve(NFF * 2);
  float*  part   = (float*)carve((size_t)32 * FFsz * 4);

  // weights -> bf16 (every call; deterministic, cheap); q|k|v concatenated
  f2bf_kernel<<<(int)((DD  + 255) / 256), 256, 0, stream>>>(wq,  wqkv_b,          (int)DD);
  f2bf_kernel<<<(int)((DD  + 255) / 256), 256, 0, stream>>>(wk,  wqkv_b + DD,     (int)DD);
  f2bf_kernel<<<(int)((DD  + 255) / 256), 256, 0, stream>>>(wv,  wqkv_b + 2 * DD, (int)DD);
  f2bf_kernel<<<(int)((DD  + 255) / 256), 256, 0, stream>>>(wo,  wo_b,  (int)DD);
  f2bf_kernel<<<(int)((DFF + 255) / 256), 256, 0, stream>>>(wup, wup_b, (int)DFF);
  f2bf_kernel<<<(int)((DFF + 255) / 256), 256, 0, stream>>>(wdn, wdn_b, (int)DFF);

  // x1 = rmsnorm(h, attn_norm_w)
  rmsnorm_cast<<<NROWS, 256, 0, stream>>>(h, anw, x_b);

  // fused qkv projection: [4096,768] x [2304,768]^T -> [4096,2304] bf16
  gemm_bf16<1><<<dim3(QKVLD / 128, NROWS / 128), 256, 0, stream>>>(
      x_b, wqkv_b, NROWS, QKVLD, Dsz, nullptr, qkv_b, nullptr, nullptr);

  // causal flash attention (reads fused qkv with stride 2304)
  flash_attn<<<dim3(Ssz / 64, Bsz * NHsz), 128, 0, stream>>>(
      qkv_b, qkv_b + Dsz, qkv_b + 2 * Dsz, o_b);

  // h1 = h + o @ wo^T   (fused residual)
  dim3 gD(Dsz / 128, NROWS / 128);  // (6, 32)
  gemm_bf16<0><<<gD, 256, 0, stream>>>(o_b, wo_b, NROWS, Dsz, Dsz, h1, nullptr, h, nullptr);

  // x2 = rmsnorm(h1, mlp_norm_w)
  rmsnorm_cast<<<NROWS, 256, 0, stream>>>(h1, mnw, x_b);

  // rate = LIF(x2 @ w_up^T, beta)  (fused epilogue)
  gemm_bf16<2><<<dim3(FFsz / 128, NROWS / 128), 256, 0, stream>>>(
      x_b, wup_b, NROWS, FFsz, Dsz, nullptr, rate_b, nullptr, beta);

  // rate_per_unit = mean over rows (deterministic two-stage)
  col_partial<<<dim3(FFsz / 256, 32), 256, 0, stream>>>(rate_b, part, NROWS / 32, FFsz);
  col_final<<<FFsz / 256, 256, 0, stream>>>(part, out + ND, 32, FFsz, 1.0f / NROWS);

  // out = h1 + rate @ w_down^T  (writes d_out directly)
  gemm_bf16<0><<<gD, 256, 0, stream>>>(rate_b, wdn_b, NROWS, Dsz, FFsz, out, nullptr, h1, nullptr);
}
